// SimpleQLoRAModel_10445360464556
// MI455X (gfx1250) — compile-verified, matrix operand order checked by reading, not run on hardware
//
#include <hip/hip_runtime.h>

// ---------------------------------------------------------------------------
// NF4-dequant + LoRA fused GEMM for MI455X (gfx1250, wave32, WMMA).
//   y = x @ (dequant_nf4(codes,absmax) + adapter)^T
//   x: [1048576, 64] f32 ; W: [64, 64] ; y: [1048576, 64] f32
// Strategy: fold the 64x64 effective weight once (prepack kernel), then run a
// memory-roofline streaming GEMM using V_WMMA_F32_16X16X32_BF16 with a
// split-precision (hi/lo bf16) scheme: acc += xh*Wh + xh*Wl + xl*Wh, giving
// ~f32 accuracy at 3x bf16-WMMA cost (still above the 23 us HBM floor).
// ---------------------------------------------------------------------------

typedef __attribute__((ext_vector_type(16))) __bf16 v16bf;
typedef __attribute__((ext_vector_type(8)))  float  v8f;
typedef __attribute__((ext_vector_type(4)))  float  v4f;

#define IN_F  64
#define OUT_F 64

__constant__ float NF4_TAB[16] = {
    -1.0f, -0.6961928009986877f, -0.5250730514526367f, -0.39491748809814453f,
    -0.28444138169288635f, -0.18477343022823334f, -0.09105003625154495f, 0.0f,
    0.07958029955625534f, 0.16093020141124725f, 0.24611230194568634f,
    0.33791524171829224f, 0.44070982933044434f, 0.5626170039176941f,
    0.7229568362236023f, 1.0f };

// Fragment element -> K mapping for 16-bit WMMA A/B operands (ISA 7.12.2):
//   lanes 0-15 : VGPR0-3 = K 0..7,  VGPR4-7 = K 16..23
//   lanes16-31 : VGPR0-3 = K 8..15, VGPR4-7 = K 24..31
// element e (0..15) of the v16bf per-lane vector:
//   K = kstep*32 + klo + (e<8 ? e : 8+e),  klo = (lane>=16)*8

// ---------------------------------------------------------------------------
// Prepack: dequantize NF4, add adapter, emit bf16 hi/lo B-fragments.
// Layout in ws: whi[frag][lane][16] then wlo[...]; frag = ntile*2 + kstep.
// One block, 256 threads: thread == (frag,lane) slot, writes 16 elems each.
// ---------------------------------------------------------------------------
__global__ __launch_bounds__(256)
void nf4_prepack_kernel(const int* __restrict__ codes,
                        const float* __restrict__ absmax,
                        const float* __restrict__ adapter,
                        __bf16* __restrict__ whi,
                        __bf16* __restrict__ wlo) {
  const int slot = threadIdx.x;        // 0..255
  const int lane = slot & 31;
  const int frag = slot >> 5;          // 0..7
  const int t    = frag >> 1;          // N-tile 0..3
  const int s    = frag & 1;           // K-step 0..1
  const int n    = t * 16 + (lane & 15);
  const int klo  = (lane & 16) ? 8 : 0;
  const float am = absmax[n];
#pragma unroll
  for (int e = 0; e < 16; ++e) {
    const int k = s * 32 + klo + ((e < 8) ? e : (8 + e));
    const float w = NF4_TAB[codes[n * IN_F + k] & 15] * am + adapter[n * IN_F + k];
    const __bf16 h = (__bf16)w;
    const int idx = (frag * 32 + lane) * 16 + e;
    whi[idx] = h;
    wlo[idx] = (__bf16)(w - (float)h);
  }
}

// ---------------------------------------------------------------------------
// Streaming GEMM: each wave32 computes 16 rows x 64 cols per tile.
// ---------------------------------------------------------------------------
__global__ __launch_bounds__(256)
void nf4_lora_gemm_wmma(const float* __restrict__ x,
                        const __bf16* __restrict__ wfrag,   // hi[4096] then lo[4096]
                        float* __restrict__ y,
                        int num_tiles) {
  const int lane       = threadIdx.x & 31;
  const int waveInBlk  = threadIdx.x >> 5;                  // 0..7
  const int wavesTotal = gridDim.x * (blockDim.x >> 5);
  const int waveId     = blockIdx.x * (blockDim.x >> 5) + waveInBlk;

  const __bf16* __restrict__ whi = wfrag;
  const __bf16* __restrict__ wlo = wfrag + 8 * 32 * 16;

  // Pin all 16 weight fragments (hi+lo) in VGPRs: 32B contiguous per lane.
  v16bf bhi[8], blo[8];
#pragma unroll
  for (int f = 0; f < 8; ++f) {
    bhi[f] = *(const v16bf*)(whi + (f * 32 + lane) * 16);
    blo[f] = *(const v16bf*)(wlo + (f * 32 + lane) * 16);
  }

  const int row   = lane & 15;               // M (and N-col) within tile
  const int klo   = (lane & 16) ? 8 : 0;     // A-fragment K base for this lane
  const int mhalf = (lane & 16) ? 8 : 0;     // C/D: lanes>=16 hold M=8..15

  for (int tile = waveId; tile < num_tiles; tile += wavesTotal) {
    const float* __restrict__ xr = x + ((size_t)tile * 16 + row) * IN_F;
    if (tile + wavesTotal < num_tiles)       // global_prefetch next tile
      __builtin_prefetch(x + ((size_t)(tile + wavesTotal) * 16 + row) * IN_F, 0, 1);

    v8f acc[4] = {};                         // 4 N-tiles of 16x16 f32

#pragma unroll
    for (int s = 0; s < 2; ++s) {            // K-steps of 32
      const float* xk = xr + s * 32 + klo;
      const v4f a0 = *(const v4f*)(xk +  0); // K = s*32+klo   .. +3
      const v4f a1 = *(const v4f*)(xk +  4); // K = s*32+klo+4 .. +7
      const v4f a2 = *(const v4f*)(xk + 16); // K = s*32+16+klo.. +3
      const v4f a3 = *(const v4f*)(xk + 20); // K = s*32+16+klo+4..+7
      const float fv[16] = { a0.x, a0.y, a0.z, a0.w, a1.x, a1.y, a1.z, a1.w,
                             a2.x, a2.y, a2.z, a2.w, a3.x, a3.y, a3.z, a3.w };
      v16bf ahi, alo;
#pragma unroll
      for (int e = 0; e < 16; ++e) {
        const __bf16 h = (__bf16)fv[e];
        ahi[e] = h;
        alo[e] = (__bf16)(fv[e] - (float)h);
      }
#pragma unroll
      for (int t = 0; t < 4; ++t) {
        const int fg = t * 2 + s;
        acc[t] = __builtin_amdgcn_wmma_f32_16x16x32_bf16(
            false, ahi, false, bhi[fg], (short)0, acc[t], false, false);
        acc[t] = __builtin_amdgcn_wmma_f32_16x16x32_bf16(
            false, ahi, false, blo[fg], (short)0, acc[t], false, false);
        acc[t] = __builtin_amdgcn_wmma_f32_16x16x32_bf16(
            false, alo, false, bhi[fg], (short)0, acc[t], false, false);
      }
    }

    // D layout (32-bit 16x16): VGPR r, lanes 0-15 -> M=r, N=lane;
    //                          lanes 16-31 -> M=8+r, N=lane-16.
    // Each b32 store covers two contiguous 64B row segments (coalesced).
    float* yr = y + ((size_t)tile * 16 + mhalf) * OUT_F + row;
#pragma unroll
    for (int t = 0; t < 4; ++t)
#pragma unroll
      for (int r = 0; r < 8; ++r)
        yr[(size_t)r * OUT_F + t * 16] = acc[t][r];
  }
}

extern "C" void kernel_launch(void* const* d_in, const int* in_sizes, int n_in,
                              void* d_out, int out_size, void* d_ws, size_t ws_size,
                              hipStream_t stream) {
  const float* x       = (const float*)d_in[0];  // [N_TOKENS, 64] f32
  const int*   codes   = (const int*)  d_in[1];  // [64, 64] int32
  const float* absmax  = (const float*)d_in[2];  // [64] f32
  const float* adapter = (const float*)d_in[3];  // [64, 64] f32
  float*       y       = (float*)d_out;

  // 16 KB of workspace: 8 KB bf16 hi-fragments + 8 KB bf16 lo-fragments.
  __bf16* wfrag = (__bf16*)d_ws;
  __bf16* whi   = wfrag;
  __bf16* wlo   = wfrag + 8 * 32 * 16;

  nf4_prepack_kernel<<<1, 256, 0, stream>>>(codes, absmax, adapter, whi, wlo);

  const int num_tiles = in_sizes[0] / (IN_F * 16);   // 65536 M-tiles of 16 rows
  // 1024 blocks * 8 waves = 8192 waves -> 8 tiles/wave (amortizes W preload).
  nf4_lora_gemm_wmma<<<1024, 256, 0, stream>>>(x, wfrag, y, num_tiles);
}
